// LocationSensitiveAttention_55886114455969
// MI455X (gfx1250) — matrix operand order repaired and checked by reading, not compiled, not run on hardware
//
#include <hip/hip_runtime.h>
#include <hip/hip_bf16.h>
#include <math.h>

// Problem constants (match reference)
#define B_   64
#define T_   1024
#define A_   128
#define F_   1024
#define K_   31
#define DQ_  1024
#define DV_  512

typedef __attribute__((ext_vector_type(16))) __bf16          v16bf;
typedef __attribute__((ext_vector_type(16))) unsigned short  v16u;
typedef __attribute__((ext_vector_type(8)))  float           v8f;
typedef __attribute__((ext_vector_type(4)))  float           v4f;
typedef __attribute__((ext_vector_type(4)))  unsigned int    v4u;

__device__ __forceinline__ unsigned short f2bf_bits(float x) {
    union { float f; unsigned u; } u; u.f = x;
    unsigned r = u.u + 0x7FFFu + ((u.u >> 16) & 1u);   // round-to-nearest-even
    return (unsigned short)(r >> 16);
}

// Branch-free tanh: v_exp_f32 + v_rcp_f32 (TRANS pipe), saturates to +/-1 correctly.
__device__ __forceinline__ float fast_tanh(float x) {
    float ex = __expf(2.f * x);
    return 1.f - 2.f * __builtin_amdgcn_rcpf(ex + 1.f);
}

// ---------------- workspace layout (bytes) ----------------
#define OFF_Q     ((size_t)0)         // q:      64*128 f32                 =   32768
#define OFF_S     ((size_t)32768)     // S:      64*64*128 f32 (62 used)    = 2097152
#define OFF_WMB   ((size_t)2129920)   // Wm  bf16 frags 16kb*8nt*512        =  131072
#define OFF_WLOCB ((size_t)2260992)   // Wloc bf16 frags 32kb*8nt*512       =  262144
#define OFF_WCA   ((size_t)2523136)   // Wc A-frags 64mt*2kb*512            =  131072
#define OFF_SBF   ((size_t)2654208)   // S bf16 frags 64b*2kb*8nt*512       = 1048576
#define OFF_V     ((size_t)3702784)   // v:      64*1024*128 f32            = 33554432
#define OFF_E     ((size_t)37257216)  // e:      64*1024 f32                =  262144
#define OFF_AL    ((size_t)37519360)  // align:  64*1024 f32                =  262144
// total ~37.8 MB

// ================= q = tanh(query @ Wq)  [B,A] =================
__global__ void k_q(const float* __restrict__ query, const float* __restrict__ Wq,
                    float* __restrict__ q) {
    int b = blockIdx.x, a = threadIdx.x;
    const float* qr = query + b * DQ_;
    float acc = 0.f;
    for (int d = 0; d < DQ_; ++d) acc += qr[d] * Wq[d * A_ + a];
    q[b * A_ + a] = fast_tanh(acc);
}

// ====== generic B-fragment swizzler: row-major f32 [K x 128] -> bf16 WMMA B frags ======
// B frag (32x16 tile): lane holds col N=lane&15; element e holds K = (lane>=16?8:0) + (e<8 ? e : e+8).
__global__ void k_bprep(const float* __restrict__ W, unsigned short* __restrict__ dstB) {
    int kb = blockIdx.x >> 3, nt = blockIdx.x & 7, lane = threadIdx.x;
    int krow = kb * 32 + ((lane >= 16) ? 8 : 0);
    int col  = nt * 16 + (lane & 15);
    unsigned short* dst = dstB + (((kb * 8 + nt) * 32 + lane) << 4);
#pragma unroll
    for (int e = 0; e < 16; ++e) dst[e] = f2bf_bits(W[(size_t)(krow + ((e < 8) ? e : (e + 8))) * A_ + col]);
}

// ====== Wconv -> A-fragments for the loc GEMM: A'[t, j] (M=t, K=j in 0..63, j>=62 zero) ======
// A frag (16x32): lane row M = lane&15; element e: K = kb*32 + (lane>=16?8:0) + (e<8 ? e : e+8)
__global__ void k_wcprep(const float* __restrict__ Wconv, unsigned short* __restrict__ wcA) {
    int mt = blockIdx.x >> 1;        // t-tile 0..63
    int kb = blockIdx.x & 1;
    int lane = threadIdx.x;
    int t = mt * 16 + (lane & 15);
    int koff = (lane >> 4) * 8;
    unsigned short* dst = wcA + (((mt * 2 + kb) * 32 + lane) << 4);
#pragma unroll
    for (int e = 0; e < 16; ++e) {
        int j = kb * 32 + koff + ((e < 8) ? e : (e + 8));
        float x = 0.f;
        if (j < 62) {
            int c = (j >= 31) ? 1 : 0;
            int k = j - c * 31;
            x = Wconv[(size_t)(k * 2 + c) * F_ + t];
        }
        dst[e] = f2bf_bits(x);
    }
}

// ====== S[b, j, a] = sum_t aw_pad[b, t+k(j)-15, c(j)] * Wloc[t,a]  via WMMA ======
// Per b: [64(j,62 used) x 1024(t)] x [1024 x 128] ; A gathered from LDS aw window, B = Wloc frags.
__global__ void __launch_bounds__(128) k_S(const float* __restrict__ aw,
                                           const unsigned short* __restrict__ wlB,
                                           float* __restrict__ S) {
    int b = blockIdx.x;
    int tid = threadIdx.x, wave = tid >> 5, lane = tid & 31;
    __shared__ float awp[2 * 1054];                       // awp[c][i] = aw[i-15], zero padded
    __shared__ __align__(32) unsigned short bsh[8 * 32 * 16];
    for (int i = tid; i < 2 * 1054; i += 128) {
        int c = i / 1054, t = (i % 1054) - 15;
        awp[i] = (t >= 0 && t < T_) ? aw[((size_t)b * T_ + t) * 2 + c] : 0.f;
    }
    int j = wave * 16 + (lane & 15);
    int valid = (j < 62);
    int c = (j >= 31) ? 1 : 0;
    int k = valid ? (j - c * 31) : 0;
    int base = (valid ? c : 0) * 1054 + k;                // awp[c][t + k]
    int koff = (lane >> 4) * 8;

    v8f acc[8];
#pragma unroll
    for (int nt = 0; nt < 8; ++nt)
#pragma unroll
        for (int r = 0; r < 8; ++r) acc[nt][r] = 0.f;

    for (int kb = 0; kb < 32; ++kb) {
        const v4u* src = (const v4u*)(wlB + kb * 4096);
        v4u* dst = (v4u*)bsh;
#pragma unroll
        for (int r = 0; r < 4; ++r) dst[tid + r * 128] = src[tid + r * 128];
        __syncthreads();

        v16u au;
#pragma unroll
        for (int e = 0; e < 16; ++e) {
            int tloc = kb * 32 + koff + ((e < 8) ? e : (e + 8));
            float x = valid ? awp[base + tloc] : 0.f;
            au[e] = f2bf_bits(x);
        }
        v16bf af = __builtin_bit_cast(v16bf, au);

        // preload ALL 8 B fragments, then issue the 8 WMMAs back-to-back
        v16u bu[8];
#pragma unroll
        for (int nt = 0; nt < 8; ++nt)
            bu[nt] = *(const v16u*)(bsh + ((nt * 32 + lane) << 4));
#pragma unroll
        for (int nt = 0; nt < 8; ++nt)
            acc[nt] = __builtin_amdgcn_wmma_f32_16x16x32_bf16(
                false, af, false, __builtin_bit_cast(v16bf, bu[nt]), (short)0, acc[nt], false, false);
        __syncthreads();
    }
    int rowAdd = (lane >= 16) ? 8 : 0, col0 = lane & 15;
#pragma unroll
    for (int nt = 0; nt < 8; ++nt)
#pragma unroll
        for (int r = 0; r < 8; ++r) {
            int jr = wave * 16 + r + rowAdd;
            S[(size_t)b * 8192 + jr * A_ + nt * 16 + col0] = acc[nt][r];
        }
}

// ====== S f32 [b][64][128] -> bf16 B-fragments (K=j dim, N=a) ======
__global__ void k_sprep(const float* __restrict__ S, unsigned short* __restrict__ Sbf) {
    int b  = blockIdx.x >> 4;
    int kb = (blockIdx.x >> 3) & 1;
    int nt = blockIdx.x & 7;
    int lane = threadIdx.x;
    int krow = kb * 32 + ((lane >= 16) ? 8 : 0);
    int col  = nt * 16 + (lane & 15);
    unsigned short* dst = Sbf + (size_t)b * 8192 + (((kb * 8 + nt) * 32 + lane) << 4);
#pragma unroll
    for (int e = 0; e < 16; ++e) dst[e] = f2bf_bits(S[(size_t)b * 8192 + (krow + ((e < 8) ? e : (e + 8))) * A_ + col]);
}

// ====== v = tanh(value @ Wm) : [65536,512]x[512,128] via v_wmma_f32_16x16x32_bf16 ======
__global__ void __launch_bounds__(128) k_v(const float* __restrict__ value,
                                           const unsigned short* __restrict__ wsB,
                                           float* __restrict__ v) {
    __shared__ __align__(32) unsigned short bsh[8 * 32 * 16];   // 8 KB
    int tid  = threadIdx.x;
    int wave = tid >> 5, lane = tid & 31;
    int mBase = blockIdx.x * 64 + wave * 16;
    const float* ap_base = value + (size_t)(mBase + (lane & 15)) * DV_;
    int koff = (lane >> 4) * 8;

    v8f acc[8];
#pragma unroll
    for (int nt = 0; nt < 8; ++nt)
#pragma unroll
        for (int r = 0; r < 8; ++r) acc[nt][r] = 0.f;

    for (int kb = 0; kb < 16; ++kb) {
        // A loads first (global latency overlaps the staging barrier)
        const float* ap = ap_base + kb * 32 + koff;
        v4f a0 = *(const v4f*)(ap);
        v4f a1 = *(const v4f*)(ap + 4);
        v4f a2 = *(const v4f*)(ap + 16);
        v4f a3 = *(const v4f*)(ap + 20);

        const v4u* src = (const v4u*)(wsB + kb * 4096);
        v4u* dst = (v4u*)bsh;
#pragma unroll
        for (int r = 0; r < 4; ++r) dst[tid + r * 128] = src[tid + r * 128];
        __syncthreads();

        v16u au;
#pragma unroll
        for (int i = 0; i < 4; ++i) {
            au[i]      = f2bf_bits(a0[i]);
            au[4 + i]  = f2bf_bits(a1[i]);
            au[8 + i]  = f2bf_bits(a2[i]);
            au[12 + i] = f2bf_bits(a3[i]);
        }
        v16bf af = __builtin_bit_cast(v16bf, au);

        // preload ALL 8 B fragments, then issue the 8 WMMAs back-to-back
        v16u bu[8];
#pragma unroll
        for (int nt = 0; nt < 8; ++nt)
            bu[nt] = *(const v16u*)(bsh + ((nt * 32 + lane) << 4));
#pragma unroll
        for (int nt = 0; nt < 8; ++nt)
            acc[nt] = __builtin_amdgcn_wmma_f32_16x16x32_bf16(
                false, af, false, __builtin_bit_cast(v16bf, bu[nt]), (short)0, acc[nt], false, false);
        __syncthreads();
    }

    int rowAdd = (lane >= 16) ? 8 : 0, col0 = lane & 15;
#pragma unroll
    for (int nt = 0; nt < 8; ++nt)
#pragma unroll
        for (int r = 0; r < 8; ++r) {
            int orow = mBase + r + rowAdd;
            v[(size_t)orow * A_ + nt * 16 + col0] = fast_tanh(acc[nt][r]);
        }
}

// ====== loc GEMM + fused score:  l2 = Wc^T(1024x62) x S[b](62x128) per b (WMMA),
//        then e[b,t] = sum_a Wv[a]*tanh(q+v+tanh(l2)) reduced in-register ======
__global__ void __launch_bounds__(128) k_locE(const unsigned short* __restrict__ wcA,
                                              const unsigned short* __restrict__ Sbf,
                                              const float* __restrict__ q,
                                              const float* __restrict__ Wv,
                                              const float* __restrict__ v,
                                              float* __restrict__ e) {
    int b     = blockIdx.x >> 4;
    int chunk = blockIdx.x & 15;
    int tid = threadIdx.x, wave = tid >> 5, lane = tid & 31;
    int mtG = chunk * 4 + wave;                     // global 16-row t tile (0..63)
    __shared__ __align__(32) unsigned short ssh[2 * 8 * 32 * 16];   // 16 KB: S frags of b
    {
        const v4u* src = (const v4u*)(Sbf + (size_t)b * 8192);
        v4u* dst = (v4u*)ssh;
#pragma unroll
        for (int r = 0; r < 8; ++r) dst[tid + r * 128] = src[tid + r * 128];
    }
    __syncthreads();

    v8f acc[8];
#pragma unroll
    for (int nt = 0; nt < 8; ++nt)
#pragma unroll
        for (int r = 0; r < 8; ++r) acc[nt][r] = 0.f;

#pragma unroll
    for (int kb = 0; kb < 2; ++kb) {
        v16u au = *(const v16u*)(wcA + (((mtG * 2 + kb) * 32 + lane) << 4));
        v16bf af = __builtin_bit_cast(v16bf, au);
        v16u bu[8];
#pragma unroll
        for (int nt = 0; nt < 8; ++nt)
            bu[nt] = *(const v16u*)(ssh + (((kb * 8 + nt) * 32 + lane) << 4));
#pragma unroll
        for (int nt = 0; nt < 8; ++nt)
            acc[nt] = __builtin_amdgcn_wmma_f32_16x16x32_bf16(
                false, af, false, __builtin_bit_cast(v16bf, bu[nt]), (short)0, acc[nt], false, false);
    }

    // fused score epilogue
    int rowAdd = (lane >= 16) ? 8 : 0, col0 = lane & 15;
    float rowsum[8];
#pragma unroll
    for (int r = 0; r < 8; ++r) rowsum[r] = 0.f;
#pragma unroll
    for (int nt = 0; nt < 8; ++nt) {
        int a = nt * 16 + col0;
        float qa = q[b * A_ + a];
        float wv = Wv[a];
#pragma unroll
        for (int r = 0; r < 8; ++r) {
            int t = mtG * 16 + r + rowAdd;
            float vv = v[((size_t)b * T_ + t) * A_ + a];
            rowsum[r] += wv * fast_tanh(qa + vv + fast_tanh(acc[nt][r]));
        }
    }
#pragma unroll
    for (int r = 0; r < 8; ++r) {
        float s = rowsum[r];
#pragma unroll
        for (int m = 8; m >= 1; m >>= 1) s += __shfl_xor(s, m, 32);
        if ((lane & 15) == 0) e[b * T_ + mtG * 16 + r + rowAdd] = s;
    }
}

// ====== softmax over t ======
__global__ void k_softmax(const float* __restrict__ e, float* __restrict__ al_ws,
                          float* __restrict__ out_al) {
    int b = blockIdx.x, tid = threadIdx.x;   // 256 threads
    __shared__ float se[T_];
    __shared__ float red[256];
#pragma unroll
    for (int i = 0; i < 4; ++i) se[tid + i * 256] = e[b * T_ + tid + i * 256];
    __syncthreads();
    float m = -1e30f;
#pragma unroll
    for (int i = 0; i < 4; ++i) m = fmaxf(m, se[tid + i * 256]);
    red[tid] = m; __syncthreads();
    for (int s = 128; s > 0; s >>= 1) { if (tid < s) red[tid] = fmaxf(red[tid], red[tid + s]); __syncthreads(); }
    m = red[0]; __syncthreads();
    float part = 0.f;
#pragma unroll
    for (int i = 0; i < 4; ++i) { float x = __expf(se[tid + i * 256] - m); se[tid + i * 256] = x; part += x; }
    red[tid] = part; __syncthreads();
    for (int s = 128; s > 0; s >>= 1) { if (tid < s) red[tid] += red[tid + s]; __syncthreads(); }
    float inv = 1.f / red[0];
#pragma unroll
    for (int i = 0; i < 4; ++i) {
        float al = se[tid + i * 256] * inv;
        al_ws[b * T_ + tid + i * 256]  = al;
        out_al[b * T_ + tid + i * 256] = al;
    }
}

// ====== context[b,a] = sum_t align[b,t] * v[b,t,a] ======
__global__ void k_ctx(const float* __restrict__ al_ws, const float* __restrict__ v,
                      float* __restrict__ out_ctx) {
    int b = blockIdx.x, a = threadIdx.x;
    __shared__ float sal[T_];
    for (int i = a; i < T_; i += 128) sal[i] = al_ws[b * T_ + i];
    __syncthreads();
    float acc = 0.f;
    for (int t = 0; t < T_; ++t) acc += sal[t] * v[((size_t)b * T_ + t) * A_ + a];
    out_ctx[b * A_ + a] = acc;
}

extern "C" void kernel_launch(void* const* d_in, const int* in_sizes, int n_in,
                              void* d_out, int out_size, void* d_ws, size_t ws_size,
                              hipStream_t stream) {
    (void)in_sizes; (void)n_in; (void)out_size; (void)ws_size;
    const float* query = (const float*)d_in[0];
    const float* value = (const float*)d_in[1];
    const float* aw    = (const float*)d_in[2];
    const float* Wq    = (const float*)d_in[3];
    const float* Wm    = (const float*)d_in[4];
    const float* Wv    = (const float*)d_in[5];
    const float* Wconv = (const float*)d_in[6];
    const float* Wloc  = (const float*)d_in[7];

    char* ws = (char*)d_ws;
    float*          q    = (float*)(ws + OFF_Q);
    float*          S    = (float*)(ws + OFF_S);
    unsigned short* wmb  = (unsigned short*)(ws + OFF_WMB);
    unsigned short* wlb  = (unsigned short*)(ws + OFF_WLOCB);
    unsigned short* wca  = (unsigned short*)(ws + OFF_WCA);
    unsigned short* sbf  = (unsigned short*)(ws + OFF_SBF);
    float*          v    = (float*)(ws + OFF_V);
    float*          e    = (float*)(ws + OFF_E);
    float*          al   = (float*)(ws + OFF_AL);

    float* out_ctx = (float*)d_out;          // context [B,A]
    float* out_al  = out_ctx + B_ * A_;      // alignment [B,1,T]

    k_q      <<<B_, A_, 0, stream>>>(query, Wq, q);
    k_bprep  <<<16 * 8, 32, 0, stream>>>(Wm, wmb);     // Wm:   K=512  -> 16 kb
    k_bprep  <<<32 * 8, 32, 0, stream>>>(Wloc, wlb);   // Wloc: K=1024 -> 32 kb
    k_wcprep <<<64 * 2, 32, 0, stream>>>(Wconv, wca);
    k_S      <<<B_, 128, 0, stream>>>(aw, wlb, S);
    k_sprep  <<<B_ * 16, 32, 0, stream>>>(S, sbf);
    k_v      <<<(B_ * T_) / 64, 128, 0, stream>>>(value, wmb, v);
    k_locE   <<<B_ * 16, 128, 0, stream>>>(wca, sbf, q, Wv, v, e);
    k_softmax<<<B_, 256, 0, stream>>>(e, al, out_al);
    k_ctx    <<<B_, A_, 0, stream>>>(al, v, out_ctx);
}
